// AdaptiveExpertSystem_4037269258997
// MI455X (gfx1250) — matrix operand order repaired and checked by reading, not compiled, not run on hardware
//
#include <hip/hip_runtime.h>
#include <math.h>

typedef __bf16 bf16_t;
typedef bf16_t v16bf __attribute__((ext_vector_type(16)));
typedef bf16_t v8bf  __attribute__((ext_vector_type(8)));
typedef float  v8f   __attribute__((ext_vector_type(8)));

static constexpr int TOK = 4096;   // B*S
static constexpr int H   = 1024;
static constexpr int I   = 4096;
static constexpr int E   = 8;
static constexpr int TM  = 128, TN = 128, TK = 32;
static constexpr int MAXPAD = 9216;  // 8192 slots + 8 experts * 128 alignment

__device__ __forceinline__ v8f wmma_bf16(v16bf a, v16bf b, v8f c) {
  return __builtin_amdgcn_wmma_f32_16x16x32_bf16(false, a, false, b, (short)0, c,
                                                 false, false);
}

__device__ __forceinline__ v16bf cat8(v8bf lo, v8bf hi) {
  return __builtin_shufflevector(lo, hi, 0, 1, 2, 3, 4, 5, 6, 7,
                                 8, 9, 10, 11, 12, 13, 14, 15);
}

// CDNA5 async DMA: global -> LDS, 16 bytes per lane, tracked by ASYNCcnt.
__device__ __forceinline__ void async_load_b128(unsigned lds_off,
                                                unsigned long long gaddr) {
  asm volatile("global_load_async_to_lds_b128 %0, %1, off"
               :: "v"(lds_off), "v"(gaddr) : "memory");
}
__device__ __forceinline__ void wait_asynccnt0() {
  asm volatile("s_wait_asynccnt 0x0" ::: "memory");
}

// ---------------- router: LN stats + logits + top2 softmax + counts ---------
__global__ __launch_bounds__(256) void k_router(
    const float* __restrict__ hs, const float* __restrict__ rlw,
    const float* __restrict__ rlb, const float* __restrict__ rW,
    const float* __restrict__ rB, float* __restrict__ xhat,
    int* __restrict__ ridx, float* __restrict__ rwgt, int* __restrict__ counts)
{
  __shared__ float red[512];
  __shared__ float stats[2];
  __shared__ float pr[8 * 256];
  const int t = blockIdx.x, tid = threadIdx.x;

  float4 x = *(const float4*)(hs + (size_t)t * H + tid * 4);
  float s  = x.x + x.y + x.z + x.w;
  float s2 = x.x * x.x + x.y * x.y + x.z * x.z + x.w * x.w;
  red[tid] = s; red[256 + tid] = s2;
  __syncthreads();
  for (int off = 128; off > 0; off >>= 1) {
    if (tid < off) { red[tid] += red[tid + off]; red[256 + tid] += red[256 + tid + off]; }
    __syncthreads();
  }
  if (tid == 0) {
    float mu  = red[0] * (1.0f / H);
    float var = red[256] * (1.0f / H) - mu * mu;
    stats[0] = mu; stats[1] = rsqrtf(var + 1e-5f);
  }
  __syncthreads();
  const float mu = stats[0], rstd = stats[1];

  float4 lw4 = *(const float4*)(rlw + tid * 4);
  float4 lb4 = *(const float4*)(rlb + tid * 4);
  float xs[4]  = {x.x, x.y, x.z, x.w};
  float lws[4] = {lw4.x, lw4.y, lw4.z, lw4.w};
  float lbs[4] = {lb4.x, lb4.y, lb4.z, lb4.w};
  float part[8] = {0, 0, 0, 0, 0, 0, 0, 0};
  #pragma unroll
  for (int j = 0; j < 4; ++j) {
    int h = tid * 4 + j;
    float xh = (xs[j] - mu) * rstd;
    xhat[(size_t)t * H + h] = xh;
    float y = fmaf(xh, lws[j], lbs[j]);
    #pragma unroll
    for (int e2 = 0; e2 < 8; ++e2)
      part[e2] = fmaf(y, rW[h * 8 + e2], part[e2]);
  }
  #pragma unroll
  for (int e2 = 0; e2 < 8; ++e2) pr[e2 * 256 + tid] = part[e2];
  __syncthreads();
  for (int off = 128; off > 0; off >>= 1) {
    if (tid < off) {
      #pragma unroll
      for (int e2 = 0; e2 < 8; ++e2) pr[e2 * 256 + tid] += pr[e2 * 256 + tid + off];
    }
    __syncthreads();
  }
  if (tid == 0) {
    float lg[8];
    #pragma unroll
    for (int e2 = 0; e2 < 8; ++e2) lg[e2] = pr[e2 * 256] + rB[e2];
    int i0 = 0;
    #pragma unroll
    for (int e2 = 1; e2 < 8; ++e2) if (lg[e2] > lg[i0]) i0 = e2;
    int i1 = (i0 == 0) ? 1 : 0;
    #pragma unroll
    for (int e2 = 0; e2 < 8; ++e2)
      if (e2 != i0 && lg[e2] > lg[i1]) i1 = e2;
    float m  = fmaxf(lg[i0], lg[i1]);
    float a0 = __expf(lg[i0] - m), a1 = __expf(lg[i1] - m);
    float inv = 1.0f / (a0 + a1);
    ridx[t * 2] = i0; ridx[t * 2 + 1] = i1;
    rwgt[t * 2] = a0 * inv; rwgt[t * 2 + 1] = a1 * inv;
    atomicAdd(&counts[i0], 1); atomicAdd(&counts[i1], 1);
  }
}

// ---------------- tiny bookkeeping kernels ----------------------------------
__global__ void k_init(int* counts) {
  if (threadIdx.x < E) counts[threadIdx.x] = 0;
}

__global__ void k_offsets(const int* __restrict__ counts, int* __restrict__ aoff,
                          int* __restrict__ cursor, int* __restrict__ totalp) {
  if (threadIdx.x == 0) {
    int off = 0;
    for (int e = 0; e < E; ++e) {
      aoff[e] = off;
      off += (counts[e] + 127) & ~127;
      cursor[e] = 0;
    }
    aoff[E] = off;
    *totalp = off;
  }
}

__global__ void k_scatter(const int* __restrict__ ridx, const int* __restrict__ aoff,
                          int* __restrict__ cursor, int* __restrict__ perm,
                          int* __restrict__ slotpos) {
  int i = blockIdx.x * blockDim.x + threadIdx.x;
  if (i >= TOK * 2) return;
  int e = ridx[i];
  int pos = atomicAdd(&cursor[e], 1) + aoff[e];
  perm[pos] = i >> 1;     // token index
  slotpos[i] = pos;
}

// ---------------- grouped GEMM 1: h = gelu(Xn @ W1 + b1) --------------------
__global__ __launch_bounds__(256) void k_ffn1(
    const float* __restrict__ xhat, const float* __restrict__ elw,
    const float* __restrict__ elb, const float* __restrict__ w1,
    const float* __restrict__ b1, const int* __restrict__ counts,
    const int* __restrict__ aoff, const int* __restrict__ totalp,
    const int* __restrict__ perm, bf16_t* __restrict__ hbuf)
{
  __shared__ __align__(16) bf16_t As[2][TM][TK + 8];   // row-major  [m][k]
  __shared__ __align__(16) bf16_t Bs[2][TN][TK + 8];   // TRANSPOSED [n][k]

  const int m0 = blockIdx.y * TM;
  if (m0 >= *totalp) return;
  const int n0 = blockIdx.x * TN;
  int e = 0;
  #pragma unroll
  for (int i = 1; i < E; ++i) if (m0 >= aoff[i]) e = i;
  const int vend = aoff[e] + counts[e];
  const float* __restrict__ W  = w1 + (size_t)e * H * I;
  const float* __restrict__ lw = elw + e * H;
  const float* __restrict__ lb = elb + e * H;

  const int tid = threadIdx.x;
  const int lane = tid & 31;
  const int wid = tid >> 5;
  const int wm = wid >> 1, wn = wid & 1;

  // Hoisted per-thread A row pointers (4 load slots, fixed row/col across K).
  const float* aptr[4];
  bool aval[4];
  #pragma unroll
  for (int it = 0; it < 4; ++it) {
    int idx = tid + it * 256;
    int r = idx >> 3, c4 = (idx & 7) << 2;
    int s = m0 + r;
    aval[it] = (s < vend);
    int tok = aval[it] ? perm[s] : 0;
    aptr[it] = xhat + (size_t)tok * H + c4;
  }
  // Hoisted per-thread B column pointers (4 slots: col n, 4 consecutive k).
  const float* bptr[4];
  #pragma unroll
  for (int it = 0; it < 4; ++it) {
    int idx = tid + it * 256;
    int n  = idx & 127;
    int kq = idx >> 7;          // 0..7 -> k = 4*kq..4*kq+3
    bptr[it] = W + (size_t)(4 * kq) * I + n0 + n;
  }

  v8f acc[2][4];
  const v8f vzero = {0.f, 0.f, 0.f, 0.f, 0.f, 0.f, 0.f, 0.f};
  #pragma unroll
  for (int mi = 0; mi < 2; ++mi)
    #pragma unroll
    for (int ni = 0; ni < 4; ++ni) acc[mi][ni] = vzero;

  float4 aReg[4];
  float  bReg[4][4];

  // global -> registers for K-slice k0
  auto stage = [&](int k0) {
    #pragma unroll
    for (int it = 0; it < 4; ++it) {
      int idx = tid + it * 256;
      int c4 = (idx & 7) << 2;
      float4 v = make_float4(0.f, 0.f, 0.f, 0.f);
      if (aval[it]) {
        float4 xv  = *(const float4*)(aptr[it] + k0);
        float4 lwv = *(const float4*)(lw + k0 + c4);
        float4 lbv = *(const float4*)(lb + k0 + c4);
        v.x = fmaf(xv.x, lwv.x, lbv.x);
        v.y = fmaf(xv.y, lwv.y, lbv.y);
        v.z = fmaf(xv.z, lwv.z, lbv.z);
        v.w = fmaf(xv.w, lwv.w, lbv.w);
      }
      aReg[it] = v;
    }
    #pragma unroll
    for (int it = 0; it < 4; ++it) {
      const float* p = bptr[it] + (size_t)k0 * I;
      bReg[it][0] = p[0];
      bReg[it][1] = p[I];
      bReg[it][2] = p[2 * I];
      bReg[it][3] = p[3 * I];
    }
  };
  // registers -> LDS buffer b (bf16 packed)
  auto commit = [&](int b) {
    #pragma unroll
    for (int it = 0; it < 4; ++it) {
      int idx = tid + it * 256;
      int r = idx >> 3, c4 = (idx & 7) << 2;
      union { bf16_t h[4]; uint2 u; } pk;
      pk.h[0] = (bf16_t)aReg[it].x; pk.h[1] = (bf16_t)aReg[it].y;
      pk.h[2] = (bf16_t)aReg[it].z; pk.h[3] = (bf16_t)aReg[it].w;
      *(uint2*)&As[b][r][c4] = pk.u;
    }
    #pragma unroll
    for (int it = 0; it < 4; ++it) {
      int idx = tid + it * 256;
      int n  = idx & 127;
      int kq = idx >> 7;
      union { bf16_t h[4]; uint2 u; } pb;
      pb.h[0] = (bf16_t)bReg[it][0]; pb.h[1] = (bf16_t)bReg[it][1];
      pb.h[2] = (bf16_t)bReg[it][2]; pb.h[3] = (bf16_t)bReg[it][3];
      *(uint2*)&Bs[b][n][4 * kq] = pb.u;
    }
  };
  // LDS buffer b -> fragments -> 8 WMMAs
  auto compute = [&](int b) {
    v16bf afrag[2], bfrag[4];
    const int ar = lane & 15;
    const int ak = (lane >> 4) << 3;    // 0 or 8
    #pragma unroll
    for (int mi = 0; mi < 2; ++mi) {
      int R = wm * 32 + mi * 16 + ar;
      v8bf lo = *(const v8bf*)&As[b][R][ak];
      v8bf hi = *(const v8bf*)&As[b][R][16 + ak];
      afrag[mi] = cat8(lo, hi);
    }
    const int bn  = lane & 15;
    const int bkk = (lane >> 4) << 3;   // k offset 0 or 8
    #pragma unroll
    for (int ni = 0; ni < 4; ++ni) {
      int C = wn * 64 + ni * 16 + bn;
      v8bf lo = *(const v8bf*)&Bs[b][C][bkk];
      v8bf hi = *(const v8bf*)&Bs[b][C][bkk + 16];
      bfrag[ni] = cat8(lo, hi);
    }
    #pragma unroll
    for (int mi = 0; mi < 2; ++mi)
      #pragma unroll
      for (int ni = 0; ni < 4; ++ni)
        acc[mi][ni] = wmma_bf16(afrag[mi], bfrag[ni], acc[mi][ni]);
  };

  // software-pipelined main loop (double-buffered LDS, 1 barrier / k-step)
  stage(0);
  commit(0);
  int cur = 0;
  for (int k0 = TK; k0 < H; k0 += TK) {
    __syncthreads();
    stage(k0);          // global loads for next slice overlap compute(cur)
    compute(cur);
    commit(cur ^ 1);
    cur ^= 1;
  }
  __syncthreads();
  compute(cur);

  const int col_l = lane & 15;
  const int rofs  = (lane >> 4) << 3;
  #pragma unroll
  for (int mi = 0; mi < 2; ++mi)
    #pragma unroll
    for (int ni = 0; ni < 4; ++ni)
      #pragma unroll
      for (int f = 0; f < 8; ++f) {
        int row = m0 + wm * 32 + mi * 16 + rofs + f;
        int col = n0 + wn * 64 + ni * 16 + col_l;
        float xv = acc[mi][ni][f] + b1[e * I + col];
        float g  = 0.5f * xv * (1.0f + erff(xv * 0.70710678118f));
        hbuf[(size_t)row * I + col] = (bf16_t)g;
      }
}

// ---------------- grouped GEMM 2: eo = h @ W2 + b2 --------------------------
__global__ __launch_bounds__(256) void k_ffn2(
    const bf16_t* __restrict__ hbuf, const float* __restrict__ w2,
    const float* __restrict__ b2, const int* __restrict__ aoff,
    const int* __restrict__ totalp, float* __restrict__ eobuf)
{
  __shared__ __align__(16) bf16_t As[2][TM][TK + 8];   // row-major  [m][k]
  __shared__ __align__(16) bf16_t Bs[2][TN][TK + 8];   // TRANSPOSED [n][k]

  const int m0 = blockIdx.y * TM;
  if (m0 >= *totalp) return;
  const int n0 = blockIdx.x * TN;
  int e = 0;
  #pragma unroll
  for (int i = 1; i < E; ++i) if (m0 >= aoff[i]) e = i;
  const float* __restrict__ W = w2 + (size_t)e * I * H;

  const int tid = threadIdx.x;
  const int lane = tid & 31;
  const int wid = tid >> 5;
  const int wm = wid >> 1, wn = wid & 1;

  // Hoisted async-DMA addresses: A tile is bf16 in global -> DMA straight to LDS.
  unsigned long long agp[2];
  unsigned alds[2][2];
  #pragma unroll
  for (int it = 0; it < 2; ++it) {
    int idx = tid + it * 256;
    int r = idx >> 2, c8 = (idx & 3) << 3;
    agp[it] = (unsigned long long)(uintptr_t)(hbuf + (size_t)(m0 + r) * I + c8);
    alds[0][it] = (unsigned)(uintptr_t)&As[0][r][c8];   // low 32 bits == LDS offset
    alds[1][it] = (unsigned)(uintptr_t)&As[1][r][c8];
  }
  const float* bptr[4];
  #pragma unroll
  for (int it = 0; it < 4; ++it) {
    int idx = tid + it * 256;
    int n  = idx & 127;
    int kq = idx >> 7;
    bptr[it] = W + (size_t)(4 * kq) * H + n0 + n;
  }

  v8f acc[2][4];
  const v8f vzero = {0.f, 0.f, 0.f, 0.f, 0.f, 0.f, 0.f, 0.f};
  #pragma unroll
  for (int mi = 0; mi < 2; ++mi)
    #pragma unroll
    for (int ni = 0; ni < 4; ++ni) acc[mi][ni] = vzero;

  float bReg[4][4];

  // A: async DMA into LDS buffer b; B: global -> registers
  auto stage = [&](int k0, int b) {
    #pragma unroll
    for (int it = 0; it < 2; ++it)
      async_load_b128(alds[b][it], agp[it] + (unsigned long long)k0 * sizeof(bf16_t));
    #pragma unroll
    for (int it = 0; it < 4; ++it) {
      const float* p = bptr[it] + (size_t)k0 * H;
      bReg[it][0] = p[0];
      bReg[it][1] = p[H];
      bReg[it][2] = p[2 * H];
      bReg[it][3] = p[3 * H];
    }
  };
  auto commit = [&](int b) {
    #pragma unroll
    for (int it = 0; it < 4; ++it) {
      int idx = tid + it * 256;
      int n  = idx & 127;
      int kq = idx >> 7;
      union { bf16_t h[4]; uint2 u; } pb;
      pb.h[0] = (bf16_t)bReg[it][0]; pb.h[1] = (bf16_t)bReg[it][1];
      pb.h[2] = (bf16_t)bReg[it][2]; pb.h[3] = (bf16_t)bReg[it][3];
      *(uint2*)&Bs[b][n][4 * kq] = pb.u;
    }
    wait_asynccnt0();    // A-tile DMA for buffer b complete
  };
  auto compute = [&](int b) {
    v16bf afrag[2], bfrag[4];
    const int ar = lane & 15;
    const int ak = (lane >> 4) << 3;
    #pragma unroll
    for (int mi = 0; mi < 2; ++mi) {
      int R = wm * 32 + mi * 16 + ar;
      v8bf lo = *(const v8bf*)&As[b][R][ak];
      v8bf hi = *(const v8bf*)&As[b][R][16 + ak];
      afrag[mi] = cat8(lo, hi);
    }
    const int bn  = lane & 15;
    const int bkk = (lane >> 4) << 3;
    #pragma unroll
    for (int ni = 0; ni < 4; ++ni) {
      int C = wn * 64 + ni * 16 + bn;
      v8bf lo = *(const v8bf*)&Bs[b][C][bkk];
      v8bf hi = *(const v8bf*)&Bs[b][C][bkk + 16];
      bfrag[ni] = cat8(lo, hi);
    }
    #pragma unroll
    for (int mi = 0; mi < 2; ++mi)
      #pragma unroll
      for (int ni = 0; ni < 4; ++ni)
        acc[mi][ni] = wmma_bf16(afrag[mi], bfrag[ni], acc[mi][ni]);
  };

  stage(0, 0);
  commit(0);
  int cur = 0;
  for (int k0 = TK; k0 < I; k0 += TK) {
    __syncthreads();
    stage(k0, cur ^ 1);   // async DMA + B loads overlap compute(cur)
    compute(cur);
    commit(cur ^ 1);
    cur ^= 1;
  }
  __syncthreads();
  compute(cur);

  const int col_l = lane & 15;
  const int rofs  = (lane >> 4) << 3;
  #pragma unroll
  for (int mi = 0; mi < 2; ++mi)
    #pragma unroll
    for (int ni = 0; ni < 4; ++ni)
      #pragma unroll
      for (int f = 0; f < 8; ++f) {
        int row = m0 + wm * 32 + mi * 16 + rofs + f;
        int col = n0 + wn * 64 + ni * 16 + col_l;
        eobuf[(size_t)row * H + col] = acc[mi][ni][f] + b2[e * H + col];
      }
}

// ---------------- weighted combine + output LayerNorm -----------------------
__global__ __launch_bounds__(256) void k_combine(
    const float* __restrict__ eobuf, const int* __restrict__ slotpos,
    const float* __restrict__ rwgt, const float* __restrict__ olw,
    const float* __restrict__ olb, float* __restrict__ out)
{
  __shared__ float red[512];
  __shared__ float stats[2];
  const int t = blockIdx.x, tid = threadIdx.x;
  const int p0 = slotpos[t * 2], p1 = slotpos[t * 2 + 1];
  const float w0 = rwgt[t * 2], w1 = rwgt[t * 2 + 1];

  float4 a = *(const float4*)(eobuf + (size_t)p0 * H + tid * 4);
  float4 b = *(const float4*)(eobuf + (size_t)p1 * H + tid * 4);
  float c0 = w0 * a.x + w1 * b.x;
  float c1 = w0 * a.y + w1 * b.y;
  float c2 = w0 * a.z + w1 * b.z;
  float c3 = w0 * a.w + w1 * b.w;

  float s  = c0 + c1 + c2 + c3;
  float s2 = c0 * c0 + c1 * c1 + c2 * c2 + c3 * c3;
  red[tid] = s; red[256 + tid] = s2;
  __syncthreads();
  for (int off = 128; off > 0; off >>= 1) {
    if (tid < off) { red[tid] += red[tid + off]; red[256 + tid] += red[256 + tid + off]; }
    __syncthreads();
  }
  if (tid == 0) {
    float mu  = red[0] * (1.0f / H);
    float var = red[256] * (1.0f / H) - mu * mu;
    stats[0] = mu; stats[1] = rsqrtf(var + 1e-5f);
  }
  __syncthreads();
  const float mu = stats[0], rstd = stats[1];

  float4 ow = *(const float4*)(olw + tid * 4);
  float4 ob = *(const float4*)(olb + tid * 4);
  float4 o;
  o.x = (c0 - mu) * rstd * ow.x + ob.x;
  o.y = (c1 - mu) * rstd * ow.y + ob.y;
  o.z = (c2 - mu) * rstd * ow.z + ob.z;
  o.w = (c3 - mu) * rstd * ow.w + ob.w;
  *(float4*)(out + (size_t)t * H + tid * 4) = o;
}

// ---------------- launcher ---------------------------------------------------
extern "C" void kernel_launch(void* const* d_in, const int* in_sizes, int n_in,
                              void* d_out, int out_size, void* d_ws, size_t ws_size,
                              hipStream_t stream) {
  (void)in_sizes; (void)n_in; (void)out_size; (void)ws_size;
  const float* hs  = (const float*)d_in[0];
  const float* rlw = (const float*)d_in[1];
  const float* rlb = (const float*)d_in[2];
  const float* rW  = (const float*)d_in[3];
  const float* rB  = (const float*)d_in[4];
  const float* elw = (const float*)d_in[5];
  const float* elb = (const float*)d_in[6];
  const float* w1  = (const float*)d_in[7];
  const float* b1  = (const float*)d_in[8];
  const float* w2  = (const float*)d_in[9];
  const float* b2  = (const float*)d_in[10];
  const float* olw = (const float*)d_in[11];
  const float* olb = (const float*)d_in[12];
  float* out = (float*)d_out;

  char* wp = (char*)d_ws;
  auto alloc = [&](size_t bytes) -> void* {
    void* p = (void*)wp;
    wp += (bytes + 255) & ~(size_t)255;
    return p;
  };
  float*  xhat    = (float*)alloc((size_t)TOK * H * sizeof(float));
  int*    ridx    = (int*)alloc((size_t)TOK * 2 * sizeof(int));
  float*  rwgt    = (float*)alloc((size_t)TOK * 2 * sizeof(float));
  int*    counts  = (int*)alloc(E * sizeof(int));
  int*    aoff    = (int*)alloc((E + 1) * sizeof(int));
  int*    cursor  = (int*)alloc(E * sizeof(int));
  int*    totalp  = (int*)alloc(sizeof(int));
  int*    perm    = (int*)alloc(MAXPAD * sizeof(int));
  int*    slotpos = (int*)alloc((size_t)TOK * 2 * sizeof(int));
  bf16_t* hbuf    = (bf16_t*)alloc((size_t)MAXPAD * I * sizeof(bf16_t));
  float*  eobuf   = (float*)alloc((size_t)MAXPAD * H * sizeof(float));

  k_init<<<1, 32, 0, stream>>>(counts);
  k_router<<<TOK, 256, 0, stream>>>(hs, rlw, rlb, rW, rB, xhat, ridx, rwgt, counts);
  k_offsets<<<1, 1, 0, stream>>>(counts, aoff, cursor, totalp);
  k_scatter<<<(TOK * 2) / 256, 256, 0, stream>>>(ridx, aoff, cursor, perm, slotpos);
  k_ffn1<<<dim3(I / TN, MAXPAD / TM), 256, 0, stream>>>(xhat, elw, elb, w1, b1,
                                                        counts, aoff, totalp, perm, hbuf);
  k_ffn2<<<dim3(H / TN, MAXPAD / TM), 256, 0, stream>>>(hbuf, w2, b2, aoff, totalp, eobuf);
  k_combine<<<TOK, 256, 0, stream>>>(eobuf, slotpos, rwgt, olw, olb, out);
}
